// AttentionTail_74225624809891
// MI455X (gfx1250) — matrix operand order repaired and checked by reading, not compiled, not run on hardware
//
#include <hip/hip_runtime.h>
#include <hip/hip_bf16.h>
#include <math.h>

// ---------------------------------------------------------------------------
// AttentionTail fused pipeline for gfx1250 (MI455X), wave32 + WMMA bf16x3.
//   C0: one-shot f32 -> bf16 hi/lo split of query, key_feats, Wq^T, Wk^T
//   K1: qproj = (query @ Wq) * SCALE        -> bf16 hi/lo planes   (WMMA)
//   K2: kproj =  key_feats @ Wk             -> bf16 hi/lo planes   (WMMA)
//   K3: attn(l in [10000,13125)) -> relu(@W2/3+b) -> dot Wm slice  (WMMA)
//       stores scalar fields s2[n,2500], s3[n,625]  (upsample+dot commute)
//   K4: attn(l in [0,10000)) -> relu(@W1+b1) -> dot Wm[0:8]
//       + bilinear(s2) + bilinear(s3) + bm -> relu -> out           (WMMA)
// All GEMM inner loops are pure dword loads + v_wmma (no cvt, no exec dance).
// ---------------------------------------------------------------------------

typedef __bf16 bf16t;
typedef bf16t v16bf __attribute__((ext_vector_type(16)));
typedef float v8f  __attribute__((ext_vector_type(8)));

constexpr int   NUM_H   = 8;
constexpr int   HDIM    = 32;
constexpr int   CDIM    = 256;
constexpr int   NQ      = 300;
constexpr int   NQ_PAD  = 304;     // 19 tiles of 16
constexpr int   LTOT    = 13125;
constexpr int   L_PAD   = 13136;   // 821 tiles of 16
constexpr int   WEDGE1  = 10000;   // 100x100
constexpr int   WEDGE2  = 12500;   // + 50x50
constexpr int   L2N     = 2500;
constexpr int   L3N     = 625;
constexpr float QSCALE  = 0.17677669529663687f;   // 32^-0.5

__device__ __forceinline__ void split_bf16(float x, unsigned short& hi, unsigned short& lo) {
  bf16t h = (bf16t)x;                       // native RNE f32->bf16
  bf16t l = (bf16t)(x - (float)h);          // residual
  hi = __builtin_bit_cast(unsigned short, h);
  lo = __builtin_bit_cast(unsigned short, l);
}
__device__ __forceinline__ int clampi(int v, int lo, int hi) {
  return v < lo ? lo : (v > hi ? hi : v);
}

union Frag { v16bf v; unsigned short u[16]; unsigned int d[8]; };

// ---------------------------------------------------------------------------
// C0: split src (f32) into bf16 hi/lo planes, zero-padding rows >= rows_valid.
// transpose != 0: src is square CDIMxCDIM, dst[c][t] = src[t][c].
// ---------------------------------------------------------------------------
__global__ void __launch_bounds__(256)
convert_split(const float* __restrict__ src,
              unsigned short* __restrict__ hi, unsigned short* __restrict__ lo,
              int rows_valid, int rows_pad, int transpose) {
  int idx = blockIdx.x * 256 + threadIdx.x;
  if (idx >= rows_pad * CDIM) return;
  int row = idx / CDIM;
  int col = idx - row * CDIM;
  float x;
  if (transpose) {
    x = src[col * CDIM + row];              // always in range (square)
  } else {
    int rc = row < rows_valid ? row : rows_valid - 1;   // clamped, unconditional load
    x = src[rc * CDIM + col];
    x = (row < rows_valid) ? x : 0.f;
  }
  unsigned short h, l;
  split_bf16(x, h, l);
  hi[idx] = h;
  lo[idx] = l;
}

// ---------------------------------------------------------------------------
// K1/K2: C = A @ B, all operands pre-split bf16 hi/lo. B is stored transposed
// ([c][t]) so both fragments load contiguous dword pairs along K.
// One wave per 16x16 tile; 3 WMMAs per K=32 chunk (bf16x3).
// ---------------------------------------------------------------------------
__global__ void __launch_bounds__(256)
proj_gemm_bf16x3(const unsigned short* __restrict__ Ahi, const unsigned short* __restrict__ Alo,
                 const unsigned short* __restrict__ BThi, const unsigned short* __restrict__ BTlo,
                 unsigned short* __restrict__ Chi, unsigned short* __restrict__ Clo,
                 int m_tiles, float out_scale) {
  const int lane = threadIdx.x & 31;
  const int wave = threadIdx.x >> 5;
  const int tile = blockIdx.x * 8 + wave;
  const int n_tiles = CDIM / 16;
  if (tile >= m_tiles * n_tiles) return;      // wave-uniform exit
  const int tm   = (tile / n_tiles) * 16;
  const int tn   = (tile % n_tiles) * 16;
  const int half = lane >> 4;
  const int lm   = lane & 15;
  const int abase = (tm + lm) * CDIM;         // A row-major [row][K]
  const int bbase = (tn + lm) * CDIM;         // B^T row-major [col][K]

  v8f acc = {0.f, 0.f, 0.f, 0.f, 0.f, 0.f, 0.f, 0.f};
  for (int kb = 0; kb < CDIM; kb += 32) {
    Frag ah, al, bh, bl;
    #pragma unroll
    for (int j = 0; j < 8; ++j) {          // A dword j: K = (2j<8?2j:2j+8)+8*half
      int K = (2 * j < 8 ? 2 * j : 2 * j + 8) + 8 * half;
      int idx = abase + kb + K;
      ah.d[j] = *(const unsigned int*)(Ahi + idx);
      al.d[j] = *(const unsigned int*)(Alo + idx);
    }
    #pragma unroll
    for (int j = 0; j < 8; ++j) {          // B dword j: K = (2j<8?2j:2j+8)+8*half
      int K = (2 * j < 8 ? 2 * j : 2 * j + 8) + 8 * half;
      int idx = bbase + kb + K;
      bh.d[j] = *(const unsigned int*)(BThi + idx);
      bl.d[j] = *(const unsigned int*)(BTlo + idx);
    }
    acc = __builtin_amdgcn_wmma_f32_16x16x32_bf16(false, ah.v, false, bh.v, (short)0, acc, false, false);
    acc = __builtin_amdgcn_wmma_f32_16x16x32_bf16(false, ah.v, false, bl.v, (short)0, acc, false, false);
    acc = __builtin_amdgcn_wmma_f32_16x16x32_bf16(false, al.v, false, bh.v, (short)0, acc, false, false);
  }
  // C layout: VGPR r -> row tm + 8*half + r, col tn + lane%16
  #pragma unroll
  for (int r = 0; r < 8; ++r) {
    int row = tm + 8 * half + r;
    int col = tn + lm;
    unsigned short h, l;
    split_bf16(acc[r] * out_scale, h, l);
    Chi[row * CDIM + col] = h;
    Clo[row * CDIM + col] = l;
  }
}

// ---------------------------------------------------------------------------
// attn tile: acc[h] = (q_tile @ k_tile^T) for 16 n-rows x 16 l-cols, 8 heads,
// bf16x3 (q already carries the 1/sqrt(d) scale). One K=32 chunk per head.
// ---------------------------------------------------------------------------
__device__ __forceinline__ void attn_tile(const unsigned short* __restrict__ qh,
                                          const unsigned short* __restrict__ ql,
                                          const unsigned short* __restrict__ kh,
                                          const unsigned short* __restrict__ kl,
                                          int tm, int tl, int lane, v8f acc[NUM_H]) {
  const int half = lane >> 4;
  const int lm   = lane & 15;
  const int qoff = (tm + lm) * CDIM;
  const int koff = (tl + lm) * CDIM;
  #pragma unroll
  for (int h = 0; h < NUM_H; ++h) {
    Frag ah, al, bh, bl;
    #pragma unroll
    for (int j = 0; j < 8; ++j) {          // A pairs: K even, (K,K+1)
      int K = (2 * j < 8 ? 2 * j : 2 * j + 8) + 8 * half;
      int idx = qoff + h * HDIM + K;
      ah.d[j] = *(const unsigned int*)(qh + idx);
      al.d[j] = *(const unsigned int*)(ql + idx);
    }
    #pragma unroll
    for (int j = 0; j < 8; ++j) {          // B pairs: K = (2j<8?2j:2j+8)+8*half
      int K = (2 * j < 8 ? 2 * j : 2 * j + 8) + 8 * half;
      int idx = koff + h * HDIM + K;
      bh.d[j] = *(const unsigned int*)(kh + idx);
      bl.d[j] = *(const unsigned int*)(kl + idx);
    }
    v8f c = {0.f, 0.f, 0.f, 0.f, 0.f, 0.f, 0.f, 0.f};
    c = __builtin_amdgcn_wmma_f32_16x16x32_bf16(false, ah.v, false, bh.v, (short)0, c, false, false);
    c = __builtin_amdgcn_wmma_f32_16x16x32_bf16(false, ah.v, false, bl.v, (short)0, c, false, false);
    c = __builtin_amdgcn_wmma_f32_16x16x32_bf16(false, al.v, false, bh.v, (short)0, c, false, false);
    acc[h] = c;
  }
}

// ---------------------------------------------------------------------------
// K3: levels 2/3. s2[n,q] = sum_h' relu(attn@W2+b2)[h'] * Wm[8+h']   (l<12500)
//                 s3[n,q] = sum_h' relu(attn@W3+b3)[h'] * Wm[16+h']  (else)
// ---------------------------------------------------------------------------
__global__ void __launch_bounds__(256)
attn_mix_lvl23(const unsigned short* __restrict__ qh, const unsigned short* __restrict__ ql,
               const unsigned short* __restrict__ kh, const unsigned short* __restrict__ kl,
               const float* __restrict__ W2, const float* __restrict__ b2,
               const float* __restrict__ W3, const float* __restrict__ b3,
               const float* __restrict__ Wm,
               float* __restrict__ s2, float* __restrict__ s3) {
  const int lane = threadIdx.x & 31;
  const int wave = threadIdx.x >> 5;
  const int tile = blockIdx.x * 8 + wave;
  const int l_tiles = (L_PAD - WEDGE1) / 16;   // 196
  const int n_tiles = NQ_PAD / 16;             // 19
  if (tile >= n_tiles * l_tiles) return;
  const int tm = (tile / l_tiles) * 16;
  const int tl = WEDGE1 + (tile % l_tiles) * 16;

  v8f acc[NUM_H];
  attn_tile(qh, ql, kh, kl, tm, tl, lane, acc);

  const int half = lane >> 4;
  const int lm   = lane & 15;
  const int l    = tl + lm;

  float s2a[8], s3a[8];
  #pragma unroll
  for (int r = 0; r < 8; ++r) { s2a[r] = 0.f; s3a[r] = 0.f; }
  #pragma unroll
  for (int hp = 0; hp < NUM_H; ++hp) {
    float w2c[NUM_H], w3c[NUM_H];
    #pragma unroll
    for (int h = 0; h < NUM_H; ++h) { w2c[h] = W2[h * NUM_H + hp]; w3c[h] = W3[h * NUM_H + hp]; }
    const float bb2 = b2[hp], bb3 = b3[hp];
    const float wm2 = Wm[NUM_H + hp], wm3 = Wm[2 * NUM_H + hp];
    #pragma unroll
    for (int r = 0; r < 8; ++r) {
      float f2v = bb2, f3v = bb3;
      #pragma unroll
      for (int h = 0; h < NUM_H; ++h) {
        float av = acc[h][r];
        f2v = fmaf(av, w2c[h], f2v);
        f3v = fmaf(av, w3c[h], f3v);
      }
      s2a[r] = fmaf(fmaxf(f2v, 0.f), wm2, s2a[r]);
      s3a[r] = fmaf(fmaxf(f3v, 0.f), wm3, s3a[r]);
    }
  }
  #pragma unroll
  for (int r = 0; r < 8; ++r) {
    int n = tm + 8 * half + r;
    if (n < NQ && l < LTOT) {              // predicated stores only
      if (l < WEDGE2) s2[n * L2N + (l - WEDGE1)] = s2a[r];
      else            s3[n * L3N + (l - WEDGE2)] = s3a[r];
    }
  }
}

// ---------------------------------------------------------------------------
// K4: level 1 + bilinear(s2, s3) + bm, final relu -> out[n, p].
// jax.image.resize bilinear = half-pixel coords with edge clamp (upsampling).
// ---------------------------------------------------------------------------
__global__ void __launch_bounds__(256)
attn_final_mask(const unsigned short* __restrict__ qh, const unsigned short* __restrict__ ql,
                const unsigned short* __restrict__ kh, const unsigned short* __restrict__ kl,
                const float* __restrict__ W1, const float* __restrict__ b1,
                const float* __restrict__ Wm, const float* __restrict__ bm,
                const float* __restrict__ s2, const float* __restrict__ s3,
                float* __restrict__ out) {
  const int lane = threadIdx.x & 31;
  const int wave = threadIdx.x >> 5;
  const int tile = blockIdx.x * 8 + wave;
  const int l_tiles = WEDGE1 / 16;   // 625
  const int n_tiles = NQ_PAD / 16;   // 19
  if (tile >= n_tiles * l_tiles) return;
  const int tm = (tile / l_tiles) * 16;
  const int tl = (tile % l_tiles) * 16;

  v8f acc[NUM_H];
  attn_tile(qh, ql, kh, kl, tm, tl, lane, acc);

  const int half = lane >> 4;
  const int lm   = lane & 15;
  const int p    = tl + lm;

  float s1a[8];
  #pragma unroll
  for (int r = 0; r < 8; ++r) s1a[r] = 0.f;
  #pragma unroll
  for (int hp = 0; hp < NUM_H; ++hp) {
    float w1c[NUM_H];
    #pragma unroll
    for (int h = 0; h < NUM_H; ++h) w1c[h] = W1[h * NUM_H + hp];
    const float bb1 = b1[hp];
    const float wm1 = Wm[hp];
    #pragma unroll
    for (int r = 0; r < 8; ++r) {
      float f = bb1;
      #pragma unroll
      for (int h = 0; h < NUM_H; ++h) f = fmaf(acc[h][r], w1c[h], f);
      s1a[r] = fmaf(fmaxf(f, 0.f), wm1, s1a[r]);
    }
  }

  const int y = p / 100, x = p % 100;
  // level 2: 50 -> 100, src = 0.5*d - 0.25
  const float fy2 = 0.5f * (float)y - 0.25f;
  const float fx2 = 0.5f * (float)x - 0.25f;
  const int y20 = (int)floorf(fy2); const float wy2 = fy2 - (float)y20;
  const int x20 = (int)floorf(fx2); const float wx2 = fx2 - (float)x20;
  const int y20c = clampi(y20, 0, 49), y21c = clampi(y20 + 1, 0, 49);
  const int x20c = clampi(x20, 0, 49), x21c = clampi(x20 + 1, 0, 49);
  // level 3: 25 -> 100, src = 0.25*d - 0.375
  const float fy3 = 0.25f * (float)y - 0.375f;
  const float fx3 = 0.25f * (float)x - 0.375f;
  const int y30 = (int)floorf(fy3); const float wy3 = fy3 - (float)y30;
  const int x30 = (int)floorf(fx3); const float wx3 = fx3 - (float)x30;
  const int y30c = clampi(y30, 0, 24), y31c = clampi(y30 + 1, 0, 24);
  const int x30c = clampi(x30, 0, 24), x31c = clampi(x30 + 1, 0, 24);

  const float bmv = bm[0];
  #pragma unroll
  for (int r = 0; r < 8; ++r) {
    int n = tm + 8 * half + r;
    int nc = n < NQ ? n : NQ - 1;          // clamped -> unconditional gathers
    const float* s2n = s2 + nc * L2N;
    float v2 = (1.f - wy2) * ((1.f - wx2) * s2n[y20c * 50 + x20c] + wx2 * s2n[y20c * 50 + x21c])
             +         wy2 * ((1.f - wx2) * s2n[y21c * 50 + x20c] + wx2 * s2n[y21c * 50 + x21c]);
    const float* s3n = s3 + nc * L3N;
    float v3 = (1.f - wy3) * ((1.f - wx3) * s3n[y30c * 25 + x30c] + wx3 * s3n[y30c * 25 + x31c])
             +         wy3 * ((1.f - wx3) * s3n[y31c * 25 + x30c] + wx3 * s3n[y31c * 25 + x31c]);
    float v = fmaxf(s1a[r] + v2 + v3 + bmv, 0.f);
    if (n < NQ) out[n * WEDGE1 + p] = v;   // predicated store only
  }
}

// ---------------------------------------------------------------------------
extern "C" void kernel_launch(void* const* d_in, const int* in_sizes, int n_in,
                              void* d_out, int out_size, void* d_ws, size_t ws_size,
                              hipStream_t stream) {
  (void)in_sizes; (void)n_in; (void)out_size; (void)ws_size;
  const float* query     = (const float*)d_in[0];
  const float* key_feats = (const float*)d_in[1];
  // d_in[2] = key_padding_mask: unused by the reference computation
  const float* Wq = (const float*)d_in[3];
  const float* Wk = (const float*)d_in[4];
  const float* W1 = (const float*)d_in[5];
  const float* b1 = (const float*)d_in[6];
  const float* W2 = (const float*)d_in[7];
  const float* b2 = (const float*)d_in[8];
  const float* W3 = (const float*)d_in[9];
  const float* b3 = (const float*)d_in[10];
  const float* Wm = (const float*)d_in[11];
  const float* bm = (const float*)d_in[12];
  float* out = (float*)d_out;

  // workspace layout (all offsets 256B-aligned: every size is a multiple of 256
  // except s2 which we round up)
  char* ws = (char*)d_ws;
  const size_t qplane = (size_t)NQ_PAD * CDIM * sizeof(unsigned short);  // 155648
  const size_t kplane = (size_t)L_PAD  * CDIM * sizeof(unsigned short);  // 6725632
  const size_t wplane = (size_t)CDIM   * CDIM * sizeof(unsigned short);  // 131072
  size_t off = 0;
  unsigned short* qAh = (unsigned short*)(ws + off); off += qplane;   // split(query)
  unsigned short* qAl = (unsigned short*)(ws + off); off += qplane;
  unsigned short* kAh = (unsigned short*)(ws + off); off += kplane;   // split(key_feats)
  unsigned short* kAl = (unsigned short*)(ws + off); off += kplane;
  unsigned short* WqTh = (unsigned short*)(ws + off); off += wplane;  // split(Wq^T)
  unsigned short* WqTl = (unsigned short*)(ws + off); off += wplane;
  unsigned short* WkTh = (unsigned short*)(ws + off); off += wplane;  // split(Wk^T)
  unsigned short* WkTl = (unsigned short*)(ws + off); off += wplane;
  unsigned short* qPh = (unsigned short*)(ws + off); off += qplane;   // q projection
  unsigned short* qPl = (unsigned short*)(ws + off); off += qplane;
  unsigned short* kPh = (unsigned short*)(ws + off); off += kplane;   // k projection
  unsigned short* kPl = (unsigned short*)(ws + off); off += kplane;
  float* s2 = (float*)(ws + off); off += ((size_t)NQ * L2N * sizeof(float) + 255) & ~(size_t)255;
  float* s3 = (float*)(ws + off); off += (size_t)NQ * L3N * sizeof(float);
  // total ~31.8 MB

  // C0: one-shot conversions
  convert_split<<<NQ_PAD, 256, 0, stream>>>(query, qAh, qAl, NQ, NQ_PAD, 0);
  convert_split<<<L_PAD, 256, 0, stream>>>(key_feats, kAh, kAl, LTOT, L_PAD, 0);
  convert_split<<<CDIM, 256, 0, stream>>>(Wq, WqTh, WqTl, CDIM, CDIM, 1);
  convert_split<<<CDIM, 256, 0, stream>>>(Wk, WkTh, WkTl, CDIM, CDIM, 1);

  // K1: q projection (scale folded in), 19*16 = 304 tiles -> 38 blocks
  proj_gemm_bf16x3<<<38, 256, 0, stream>>>(qAh, qAl, WqTh, WqTl, qPh, qPl, NQ_PAD / 16, QSCALE);
  // K2: k projection, 821*16 = 13136 tiles -> 1642 blocks
  proj_gemm_bf16x3<<<1642, 256, 0, stream>>>(kAh, kAl, WkTh, WkTl, kPh, kPl, L_PAD / 16, 1.0f);
  // K3: levels 2/3 -> s2, s3 : 19 * 196 = 3724 tiles -> 466 blocks
  attn_mix_lvl23<<<466, 256, 0, stream>>>(qPh, qPl, kPh, kPl, W2, b2, W3, b3, Wm, s2, s3);
  // K4: level 1 + upsample-combine -> out : 19 * 625 = 11875 tiles -> 1485 blocks
  attn_final_mask<<<1485, 256, 0, stream>>>(qPh, qPl, kPh, kPl, W1, b1, Wm, bm, s2, s3, out);
}
